// AutoIntMLP_50422916055575
// MI455X (gfx1250) — compile-verified, matrix operand order checked by reading, not run on hardware
//
#include <hip/hip_runtime.h>
#include <hip/hip_bf16.h>

typedef __bf16 v16bf __attribute__((ext_vector_type(16)));
typedef float  v8f   __attribute__((ext_vector_type(8)));

#define NF 24         // fields (tokens)
#define ED 128        // embed dim
#define NH 2          // heads
#define DH 64         // head dim
#define NL 3          // attention layers
#define SPB 2         // samples per block
#define RR (SPB*NF)   // 48 valid token rows
#define RP 56         // padded token rows (guard rows for 16-tile reads)
#define SX 136        // bf16 stride: residual copy
#define SQ 264        // bf16 stride: Q|K buffer (2*128 + 8)
#define SVT 56        // bf16 stride: transposed V (token dim, padded)
#define SW 136        // bf16 stride: transposed weight (fallback path)
#define PSTR 40       // bf16 stride: probs rows (32 K + pad)
#define NT 256        // 8 wave32
#define WT_BYTES (9u * ED * ED * sizeof(__bf16))   // 294912

// One-shot: W[l,w][k][n] fp32 -> Wt[(l*3+w)][n][k] bf16 (row-major over k)
__launch_bounds__(NT)
__global__ void prep_weights(const float* __restrict__ Wq,
                             const float* __restrict__ Wk,
                             const float* __restrict__ Wv,
                             __bf16* __restrict__ Wt)
{
    int idx = blockIdx.x * NT + threadIdx.x;        // 9*128*128 total
    int m = idx / (ED * ED);                        // matrix 0..8
    int r = idx % (ED * ED);
    int l = m / 3, w = m % 3;
    int k = r >> 7, n = r & 127;
    const float* W = (w == 0 ? Wq : (w == 1 ? Wk : Wv));
    Wt[(m * ED + n) * ED + k] = (__bf16)W[l * ED * ED + r];
}

__launch_bounds__(NT)
__global__ void autoint_fused(const int* __restrict__ x,
                              const float* __restrict__ emb,
                              const float* __restrict__ Wq,
                              const float* __restrict__ Wk,
                              const float* __restrict__ Wv,
                              const float* __restrict__ fw,
                              const float* __restrict__ W1,
                              const float* __restrict__ b1,
                              const float* __restrict__ W2,
                              const float* __restrict__ b2,
                              const float* __restrict__ W3,
                              const float* __restrict__ b3,
                              const __bf16* __restrict__ Wt,   // may be null
                              float* __restrict__ out)
{
    __shared__ float  sAtt[RR * ED];               // 24.0 KB residual (fp32)
    __shared__ __bf16 sX[RR * SX];                 // 12.8 KB bf16 residual copy (A)
    __shared__ __bf16 sQKV[RP * SQ];               // 28.9 KB Q|K bf16 (+guard rows)
    __shared__ __bf16 sVt[ED * SVT];               // 14.0 KB V transposed [e][r]
    __shared__ __bf16 sWt[ED * SW];                // 34.0 KB Wt (fallback only)
    __shared__ float  sSc[SPB * NH * 32 * 32];     // 16.0 KB padded score tiles
    __shared__ __bf16 sPb[SPB * NH * 32 * PSTR];   // 10.0 KB probs bf16 (zero-padded)
    __shared__ float  sH1[SPB * 32];
    __shared__ float  sH2[SPB * 32];
    __shared__ float  sDnn[SPB];
    __shared__ float  sRed[NT];
    __shared__ int    sIds[RR];

    const int t     = threadIdx.x;
    const int bBase = blockIdx.x * SPB;

    // ---- phase 0: ids + zero all pad regions consumed by WMMA K-dims ----
    if (t < RR) {
        int b = t / NF, f = t % NF;
        sIds[t] = x[(bBase + b) * NF + f] + f * 1000;
    }
    for (int i = t; i < SPB * NH * 32 * PSTR; i += NT) sPb[i] = (__bf16)0.f;
    for (int i = t; i < ED * (RP - RR); i += NT) {          // sVt guard columns
        int e = i >> 3, r = RR + (i & 7);
        sVt[e * SVT + r] = (__bf16)0.f;
    }
    for (int i = t; i < (RP - RR) * SQ; i += NT)            // sQKV guard rows
        sQKV[RR * SQ + i] = (__bf16)0.f;
    __syncthreads();

    // ---- phase 1: gather embeddings -> sAtt (f32) + sX (bf16) ----
    for (int i = t; i < RR * ED; i += NT) {
        int r = i >> 7, c = i & 127;
        float v = emb[(long)sIds[r] * ED + c];
        sAtt[i] = v;
        sX[r * SX + c] = (__bf16)v;
    }
    __syncthreads();

    // ---- phase 2: DNN branch (on e, split-K x4 over all 256 lanes) ----
    {
        int b = t >> 7, j = (t >> 2) & 31, sp = t & 3;
        const float* e = &sAtt[b * NF * ED];
        float acc = 0.f;
        int i0 = sp * 768;
        for (int i = 0; i < 768; ++i)
            acc = fmaf(e[i0 + i], W1[(i0 + i) * 32 + j], acc);
        sRed[t] = acc;
    }
    __syncthreads();
    if (t < SPB * 32) {
        int b = t >> 5, j = t & 31;
        int p0 = (b << 7) + (j << 2);
        float acc = b1[j] + sRed[p0] + sRed[p0 + 1] + sRed[p0 + 2] + sRed[p0 + 3];
        sH1[t] = acc > 0.f ? acc : 0.f;
    }
    __syncthreads();
    if (t < SPB * 32) {
        int b = t >> 5, j = t & 31;
        float acc = b2[j];
        for (int k = 0; k < 32; ++k)
            acc = fmaf(sH1[b * 32 + k], W2[k * 32 + j], acc);
        sH2[t] = acc > 0.f ? acc : 0.f;
    }
    __syncthreads();
    if (t < SPB) {
        float acc = b3[0];
        for (int k = 0; k < 32; ++k)
            acc = fmaf(sH2[t * 32 + k], W3[k], acc);
        sDnn[t] = 1.f / (1.f + __expf(-acc));
    }

    const int wave = t >> 5;
    const int lane = t & 31;
    const int ln   = lane & 15;   // row/col within 16-tile
    const int hs   = lane >> 4;   // K-half select per 16-bit A/B VGPR layout

    // load one 16x32 bf16 operand: two contiguous 16B chunks per lane
    auto ldop = [&](const __bf16* base) -> v16bf {
        v16bf v;
#pragma unroll
        for (int i = 0; i < 8; ++i) { v[i] = base[i]; v[8 + i] = base[16 + i]; }
        return v;
    };

    // ---- phase 3: attention layers ----
    for (int l = 0; l < NL; ++l) {
        // Q/K/V projections: [48x128] @ [128x128] each, WMMA bf16.
        // Each wave owns N-tile nT == wave: its B column-tile (K=128) is
        // loop-invariant, loaded ONCE into 32 VGPRs, then 3 M-tiles stream
        // A from LDS through 4 chained WMMAs each.
        const int nT = wave;
        for (int w = 0; w < 3; ++w) {
            __syncthreads();
            v16bf B0, B1, B2, B3;
            if (Wt) {
                // fast path: pre-transposed bf16 weights straight from global/L2
                const __bf16* Wg =
                    &Wt[((l * 3 + w) * ED + nT * 16 + ln) * ED + hs * 8];
                B0 = ldop(Wg +  0);
                B1 = ldop(Wg + 32);
                B2 = ldop(Wg + 64);
                B3 = ldop(Wg + 96);
            } else {
                // fallback: stage W^T bf16 into LDS (uniform branch, whole block)
                const float* Ws = (w == 0 ? Wq : (w == 1 ? Wk : Wv)) + l * ED * ED;
                for (int i = t; i < ED * ED; i += NT) {
                    int k = i >> 7, n = i & 127;
                    sWt[n * SW + k] = (__bf16)Ws[i];
                }
                __syncthreads();
                const __bf16* Wl = &sWt[(nT * 16 + ln) * SW + hs * 8];
                B0 = ldop(Wl +  0);
                B1 = ldop(Wl + 32);
                B2 = ldop(Wl + 64);
                B3 = ldop(Wl + 96);
            }
            for (int mT = 0; mT < 3; ++mT) {
                const __bf16* pa = &sX[(mT * 16 + ln) * SX + hs * 8];
                v8f acc = {0.f, 0.f, 0.f, 0.f, 0.f, 0.f, 0.f, 0.f};
                acc = __builtin_amdgcn_wmma_f32_16x16x32_bf16(
                    false, ldop(pa +  0), false, B0, (short)0, acc, false, false);
                acc = __builtin_amdgcn_wmma_f32_16x16x32_bf16(
                    false, ldop(pa + 32), false, B1, (short)0, acc, false, false);
                acc = __builtin_amdgcn_wmma_f32_16x16x32_bf16(
                    false, ldop(pa + 64), false, B2, (short)0, acc, false, false);
                acc = __builtin_amdgcn_wmma_f32_16x16x32_bf16(
                    false, ldop(pa + 96), false, B3, (short)0, acc, false, false);
                if (w < 2) {                              // Q,K row-major
#pragma unroll
                    for (int i = 0; i < 8; ++i) {
                        int row = mT * 16 + hs * 8 + i;
                        sQKV[row * SQ + w * ED + nT * 16 + ln] = (__bf16)acc[i];
                    }
                } else {                                  // V transposed [e][r]
                    int e0 = nT * 16 + ln;
#pragma unroll
                    for (int i = 0; i < 8; ++i)
                        sVt[e0 * SVT + mT * 16 + hs * 8 + i] = (__bf16)acc[i];
                }
            }
        }
        __syncthreads();

        // scores = Q.K^T per (b,h): 2x2 padded 16-tiles, K=64 (2 steps)
        for (int tt = wave; tt < 16; tt += 8) {
            int b = tt >> 3, h = (tt >> 2) & 1, mT = (tt >> 1) & 1, nTs = tt & 1;
            v8f acc = {0.f, 0.f, 0.f, 0.f, 0.f, 0.f, 0.f, 0.f};
#pragma unroll
            for (int kb = 0; kb < DH; kb += 32) {
                v16bf a  = ldop(&sQKV[(b * NF + mT * 16 + ln) * SQ + h * DH + kb + hs * 8]);
                v16bf bm = ldop(&sQKV[(b * NF + nTs * 16 + ln) * SQ + ED + h * DH + kb + hs * 8]);
                acc = __builtin_amdgcn_wmma_f32_16x16x32_bf16(
                    false, a, false, bm, (short)0, acc, false, false);
            }
            int bh = b * NH + h;
#pragma unroll
            for (int i = 0; i < 8; ++i)
                sSc[bh * 1024 + (mT * 16 + hs * 8 + i) * 32 + nTs * 16 + ln] = acc[i];
        }
        __syncthreads();

        // softmax over valid 24x24 region -> bf16 probs (K-pad stays zero)
        if (t < SPB * NH * NF) {
            int bh = t / NF, f = t % NF;
            float* row = &sSc[bh * 1024 + f * 32];
            float m = row[0];
            for (int g = 1; g < NF; ++g) m = fmaxf(m, row[g]);
            float s = 0.f;
            float e2[NF];
            for (int g = 0; g < NF; ++g) { e2[g] = __expf(row[g] - m); s += e2[g]; }
            float inv = 1.f / s;
            __bf16* pr = &sPb[(bh * 32 + f) * PSTR];
            for (int g = 0; g < NF; ++g) pr[g] = (__bf16)(e2[g] * inv);
        }
        __syncthreads();

        // out = probs @ V via WMMA (single K=32 step, zero-padded), residual add
        for (int tt = wave; tt < 32; tt += 8) {
            int b = tt >> 4, h = (tt >> 3) & 1, mT = (tt >> 2) & 1, nTp = tt & 3;
            int bh = b * NH + h;
            v16bf a  = ldop(&sPb[(bh * 32 + mT * 16 + ln) * PSTR + hs * 8]);
            v16bf bm = ldop(&sVt[(h * DH + nTp * 16 + ln) * SVT + b * NF + hs * 8]);
            v8f acc = {0.f, 0.f, 0.f, 0.f, 0.f, 0.f, 0.f, 0.f};
            acc = __builtin_amdgcn_wmma_f32_16x16x32_bf16(
                false, a, false, bm, (short)0, acc, false, false);
            int e0 = h * DH + nTp * 16 + ln;
#pragma unroll
            for (int i = 0; i < 8; ++i) {
                int f = mT * 16 + hs * 8 + i;
                if (f < NF) sAtt[(b * NF + f) * ED + e0] += acc[i];
            }
        }
        __syncthreads();

        // refresh bf16 residual copy for next layer's projections
        if (l < NL - 1) {
            for (int i = t; i < RR * ED; i += NT) {
                int r = i >> 7, c = i & 127;
                sX[r * SX + c] = (__bf16)sAtt[i];
            }
        }
    }
    __syncthreads();

    // ---- phase 4: att_out = att . final_w ; fuse with DNN sigmoid ----
    {
        int b = t >> 7, i = t & 127;
        float p = 0.f;
        for (int j = 0; j < NF; ++j)
            p = fmaf(sAtt[b * NF * ED + j * ED + i], fw[j * ED + i], p);
        sRed[t] = p;
    }
    __syncthreads();
    if (t < SPB) {
        float s = 0.f;
        for (int k = 0; k < 128; ++k) s += sRed[t * 128 + k];
        float y = s + sDnn[t];
        out[bBase + t] = 1.f / (1.f + __expf(-y));
    }
}

extern "C" void kernel_launch(void* const* d_in, const int* in_sizes, int n_in,
                              void* d_out, int out_size, void* d_ws, size_t ws_size,
                              hipStream_t stream) {
    const int*   x   = (const int*)  d_in[0];
    const float* emb = (const float*)d_in[1];
    const float* Wq  = (const float*)d_in[2];
    const float* Wk  = (const float*)d_in[3];
    const float* Wv  = (const float*)d_in[4];
    const float* fw  = (const float*)d_in[5];
    const float* W1  = (const float*)d_in[6];
    const float* b1  = (const float*)d_in[7];
    const float* W2  = (const float*)d_in[8];
    const float* b2  = (const float*)d_in[9];
    const float* W3  = (const float*)d_in[10];
    const float* b3  = (const float*)d_in[11];
    float* outp = (float*)d_out;

    __bf16* Wt = nullptr;
    if (d_ws && ws_size >= WT_BYTES) {
        Wt = (__bf16*)d_ws;
        int total = 9 * ED * ED;                 // 147456 elements
        prep_weights<<<total / NT, NT, 0, stream>>>(Wq, Wk, Wv, Wt);
    }

    int B    = out_size;       // 16384 samples
    int grid = B / SPB;        // 8192 blocks of 256 threads
    autoint_fused<<<grid, NT, 0, stream>>>(x, emb, Wq, Wk, Wv, fw,
                                           W1, b1, W2, b2, W3, b3, Wt, outp);
}
